// RTDETRTransformerv2_14860586844349
// MI455X (gfx1250) — compile-verified, matrix operand order checked by reading, not run on hardware
//
#include <hip/hip_runtime.h>
#include <hip/hip_bf16.h>

// ---------------------------------------------------------------------------
// Problem constants (from the reference)
// ---------------------------------------------------------------------------
#define EMBED_DIM 256
#define NUM_HEADS 8
#define HEAD_DIM 32
#define PT 16
#define BS 8
#define LQ 1000
#define LV 21760          // 128*128 + 64*64 + 32*32 + 16*16
#define NQ (BS * LQ)      // 8000
#define MV (BS * LV)      // 174080
#define KDIM 256          // all GEMMs have K = 256
#define KT32 (KDIM / 32)  // 8 K-steps

// CDNA5 WMMA vector types
typedef __attribute__((ext_vector_type(16))) __bf16 v16bf;
typedef __attribute__((ext_vector_type(8)))  float  v8f;
typedef __attribute__((ext_vector_type(4)))  float  v4f;

union BF16x16 { v16bf v; unsigned p[8]; unsigned short u[16]; };

__device__ __forceinline__ unsigned short f32_to_bf16_rne(float f) {
    union { float f; unsigned u; } x; x.f = f;
    unsigned u = x.u + 0x7FFFu + ((x.u >> 16) & 1u);
    return (unsigned short)(u >> 16);
}

// Truncating pack of two f32 into one dword of two bf16 (lo in [15:0]).
// Lowers to v_mov_b16 + v_and_or_b32 (2 VALU per pair).
__device__ __forceinline__ unsigned pack_bf16_pair(float lo, float hi) {
    union { float f; unsigned u; } a, b; a.f = lo; b.f = hi;
    return (a.u >> 16) | (b.u & 0xFFFF0000u);
}

// ---------------------------------------------------------------------------
// Pack a KxN f32 weight matrix into WMMA bf16 B-fragment layout:
//   out[((nt16*(K/32) + kt)*32 + lane)*16 + i]
// For lane L, element i (ISA 7.12.2, 16-bit B 32x16):
//   n_local = L%16 ; k_local = (L>=16?8:0) + (i>=8?16:0) + (i%8)
// ---------------------------------------------------------------------------
__global__ void pack_b_bf16_kernel(const float* __restrict__ W,
                                   unsigned short* __restrict__ out,
                                   int K, int N) {
    int idx = blockIdx.x * blockDim.x + threadIdx.x;
    if (idx >= K * N) return;
    int i    = idx & 15;
    int lane = (idx >> 4) & 31;
    int t    = idx >> 9;                 // nt16*(K/32) + kt
    int KT   = K >> 5;
    int kt   = t % KT;
    int nt16 = t / KT;
    int kl = ((lane >= 16) ? 8 : 0) + ((i >= 8) ? 16 : 0) + (i & 7);
    int nl = lane & 15;
    out[idx] = f32_to_bf16_rne(W[(size_t)(kt * 32 + kl) * N + nt16 * 16 + nl]);
}

// ---------------------------------------------------------------------------
// bf16 WMMA GEMM: out[M, NH*N16*16] = A[M,256](f32) * Bpacked + bias
// 256 threads = 8 waves. Each wave computes a 16 x (N16*16) slice; NH waves
// (adjacent in the block -> A rows shared through WGP$) cover one row strip.
// N16 = 8 keeps accumulators at 64 VGPRs: no scratch spills in the K-loop
// (the 16-accumulator variant spilled), while each A fragment feeds 8 WMMAs.
// permuteValue != 0: write out in [b][h][Lv][hd] layout (value projection).
// ---------------------------------------------------------------------------
template <int N16, int NH>
__global__ void gemm_bf16_wmma_kernel(const float* __restrict__ A,
                                      const unsigned short* __restrict__ Bp,
                                      const float* __restrict__ bias,
                                      float* __restrict__ out,
                                      int M, int permuteValue) {
    const int lane = threadIdx.x & 31;
    const int wave = threadIdx.x >> 5;
    const int gw   = blockIdx.x * 8 + wave;
    const int mT   = gw / NH;                  // 16-row strip
    const int half = gw % NH;                  // which N16*16-wide slice
    if ((mT << 4) >= M) return;

    const int mBase = mT << 4;
    const int mRow  = mBase + (lane & 15);
    const int kHalf = (lane >> 4) << 3;        // 0 or 8

    v8f acc[N16];
    #pragma unroll
    for (int j = 0; j < N16; ++j) acc[j] = (v8f){};

    const float* aRow = A + (size_t)mRow * KDIM + kHalf;
    const unsigned short* BpH = Bp + (size_t)half * N16 * (KT32 * 32 * 16);

    #pragma unroll
    for (int kt = 0; kt < KT32; ++kt) {
        // ---- A fragment: two contiguous 32B chunks per lane, f32 -> bf16 ----
        const float* aPtr = aRow + (kt << 5);
        v4f a0 = *(const v4f*)(aPtr + 0);
        v4f a1 = *(const v4f*)(aPtr + 4);
        v4f a2 = *(const v4f*)(aPtr + 16);
        v4f a3 = *(const v4f*)(aPtr + 20);
        if (kt + 1 < KT32) __builtin_prefetch(aPtr + 32, 0, 0);
        BF16x16 af;
        af.p[0] = pack_bf16_pair(a0[0], a0[1]);
        af.p[1] = pack_bf16_pair(a0[2], a0[3]);
        af.p[2] = pack_bf16_pair(a1[0], a1[1]);
        af.p[3] = pack_bf16_pair(a1[2], a1[3]);
        af.p[4] = pack_bf16_pair(a2[0], a2[1]);
        af.p[5] = pack_bf16_pair(a2[2], a2[3]);
        af.p[6] = pack_bf16_pair(a3[0], a3[1]);
        af.p[7] = pack_bf16_pair(a3[2], a3[3]);

        // ---- B fragments: prepacked, contiguous 32B per lane (WGP$-hot) ----
        const unsigned short* bKt = BpH + ((size_t)kt * 32 + lane) * 16;
        #pragma unroll
        for (int j = 0; j < N16; ++j) {
            const v16bf b = *(const v16bf*)(bKt + (size_t)j * (KT32 * 32 * 16));
            acc[j] = __builtin_amdgcn_wmma_f32_16x16x32_bf16(
                false, af.v, false, b, (short)0, acc[j], false, false);
        }
    }

    // ---- store D: VGPR r holds M = r (+8 for lanes 16..31), N = lane%16 ----
    const int mOff = mBase + ((lane >> 4) << 3);
    const int nCol = lane & 15;
    const int N    = NH * N16 * 16;
    const int cBase = half * (N16 * 16);
    #pragma unroll
    for (int j = 0; j < N16; ++j) {
        const int col = cBase + (j << 4) + nCol;
        const float bv = bias[col];
        #pragma unroll
        for (int r = 0; r < 8; ++r) {
            const int row = mOff + r;
            const float val = acc[j][r] + bv;
            if (!permuteValue) {
                out[(size_t)row * N + col] = val;
            } else {
                // row = b*LV + l ; col = h*32 + d  ->  [b][h][LV][32]
                const int b = row / LV, l = row % LV;
                const int h = col >> 5, d = col & 31;
                out[(((size_t)(b * NUM_HEADS + h)) * LV + l) * HEAD_DIM + d] = val;
            }
        }
    }
}

// ---------------------------------------------------------------------------
// Deformable sampling: one wave per (b, q, h), lane = channel (HEAD_DIM=32).
// v is [b][h][LV][32] (178 MB -> resident in MI455X's 192 MB L2, so the
// random bilinear gathers hit L2, not HBM). Corner reads are 128B coalesced.
// ---------------------------------------------------------------------------
__global__ void ms_deform_sample_kernel(const float* __restrict__ ref,   // [8,1000,1,4]
                                        const float* __restrict__ v,     // [8][8][LV][32]
                                        const float* __restrict__ qoff,  // [8000][256]
                                        const float* __restrict__ qattn, // [8000][128]
                                        float* __restrict__ feats) {     // [8000][256]
    const int gwave = (blockIdx.x * blockDim.x + threadIdx.x) >> 5;
    const int lane  = threadIdx.x & 31;
    if (gwave >= NQ * NUM_HEADS) return;
    const int h   = gwave & 7;
    const int row = gwave >> 3;          // b*LQ + q
    const int b   = row / LQ;

    const v4f rp = *(const v4f*)(ref + (size_t)row * 4);  // cx, cy, w, h

    // softmax over the 16 attention logits (uniform across lanes)
    const float* al = qattn + (size_t)row * 128 + h * 16;
    float logit[PT];
    float mx = -1e30f;
    #pragma unroll
    for (int p = 0; p < PT; ++p) { logit[p] = al[p]; mx = fmaxf(mx, logit[p]); }
    float aw[PT]; float sum = 0.0f;
    #pragma unroll
    for (int p = 0; p < PT; ++p) { aw[p] = __expf(logit[p] - mx); sum += aw[p]; }
    const float inv = 1.0f / sum;

    const float* op    = qoff + (size_t)row * 256 + h * 32;   // 16 x (dx,dy)
    const float* vbase = v + ((size_t)(b * NUM_HEADS + h)) * LV * HEAD_DIM;

    const int   dims[4]   = {128, 64, 32, 16};
    const int   vstart[4] = {0, 16384, 20480, 21504};
    const float npScale   = 0.25f * 0.5f;   // (1/num_points) * OFFSET_SCALE

    float acc = 0.0f;
    int p = 0;
    #pragma unroll
    for (int l = 0; l < 4; ++l) {
        const int   HW = dims[l];
        const float fHW = (float)HW;
        const float* lvl = vbase + (size_t)vstart[l] * HEAD_DIM;
        #pragma unroll
        for (int pp = 0; pp < 4; ++pp, ++p) {
            const float lx = rp.x + op[p * 2 + 0] * npScale * rp.z;
            const float ly = rp.y + op[p * 2 + 1] * npScale * rp.w;
            const float x  = lx * fHW - 0.5f;
            const float y  = ly * fHW - 0.5f;
            const float x0f = floorf(x), y0f = floorf(y);
            const float fx = x - x0f, fy = y - y0f;
            const int x0 = (int)x0f, y0 = (int)y0f;
            const float wp = aw[p] * inv;

            #pragma unroll
            for (int c = 0; c < 4; ++c) {
                const int xi = x0 + (c & 1);
                const int yi = y0 + (c >> 1);
                const float wx = (c & 1) ? fx : (1.0f - fx);
                const float wy = (c >> 1) ? fy : (1.0f - fy);
                const bool valid = (xi >= 0) & (xi < HW) & (yi >= 0) & (yi < HW);
                const int xc = min(max(xi, 0), HW - 1);
                const int yc = min(max(yi, 0), HW - 1);
                const float w = valid ? (wx * wy * wp) : 0.0f;
                acc += w * lvl[(size_t)(yc * HW + xc) * HEAD_DIM + lane];
            }
        }
    }
    feats[(size_t)row * EMBED_DIM + h * HEAD_DIM + lane] = acc;
}

// ---------------------------------------------------------------------------
// Launcher
// ---------------------------------------------------------------------------
extern "C" void kernel_launch(void* const* d_in, const int* in_sizes, int n_in,
                              void* d_out, int out_size, void* d_ws, size_t ws_size,
                              hipStream_t stream) {
    const float* query  = (const float*)d_in[0];
    const float* refpts = (const float*)d_in[1];
    const float* value  = (const float*)d_in[2];
    // d_in[3] = spatial_shapes (compile-time known)
    const float* w_off  = (const float*)d_in[4];
    const float* b_off  = (const float*)d_in[5];
    const float* w_attn = (const float*)d_in[6];
    const float* b_attn = (const float*)d_in[7];
    const float* w_val  = (const float*)d_in[8];
    const float* b_val  = (const float*)d_in[9];
    const float* w_out  = (const float*)d_in[10];
    const float* b_out  = (const float*)d_in[11];
    float* out = (float*)d_out;

    // ---- workspace layout ----
    char* ws = (char*)d_ws;
    unsigned short* pk_wval  = (unsigned short*)(ws);                        // 128 KB
    unsigned short* pk_woff  = (unsigned short*)(ws + (128u << 10));         // 128 KB
    unsigned short* pk_wattn = (unsigned short*)(ws + (256u << 10));         //  64 KB
    unsigned short* pk_wout  = (unsigned short*)(ws + (320u << 10));         // 128 KB
    float* v_proj = (float*)(ws + (448u << 10));                             // 178.3 MB
    size_t off1   = (448u << 10) + (size_t)MV * EMBED_DIM * sizeof(float);
    float* q_off  = (float*)(ws + off1);                                     // 8.19 MB
    size_t off2   = off1 + (size_t)NQ * 256 * sizeof(float);
    float* q_attn = (float*)(ws + off2);                                     // 4.10 MB
    size_t off3   = off2 + (size_t)NQ * 128 * sizeof(float);
    float* feats  = (float*)(ws + off3);                                     // 8.19 MB

    // ---- 1) pack the four weight matrices into WMMA B-fragment layout ----
    pack_b_bf16_kernel<<<(256 * 256 + 255) / 256, 256, 0, stream>>>(w_val,  pk_wval,  256, 256);
    pack_b_bf16_kernel<<<(256 * 256 + 255) / 256, 256, 0, stream>>>(w_off,  pk_woff,  256, 256);
    pack_b_bf16_kernel<<<(256 * 128 + 255) / 256, 256, 0, stream>>>(w_attn, pk_wattn, 256, 128);
    pack_b_bf16_kernel<<<(256 * 256 + 255) / 256, 256, 0, stream>>>(w_out,  pk_wout,  256, 256);

    // ---- 2) value projection (memory-bound, head-major permuted output) ----
    gemm_bf16_wmma_kernel<8, 2><<<((MV / 16) * 2 + 7) / 8, 256, 0, stream>>>(
        value, pk_wval, b_val, v_proj, MV, 1);

    // ---- 3) offset + attention projections ----
    gemm_bf16_wmma_kernel<8, 2><<<((NQ / 16) * 2 + 7) / 8, 256, 0, stream>>>(
        query, pk_woff, b_off, q_off, NQ, 0);
    gemm_bf16_wmma_kernel<8, 1><<<((NQ / 16) + 7) / 8, 256, 0, stream>>>(
        query, pk_wattn, b_attn, q_attn, NQ, 0);

    // ---- 4) softmax + bilinear sampling + point reduction ----
    ms_deform_sample_kernel<<<NQ, 256, 0, stream>>>(refpts, v_proj, q_off, q_attn, feats);

    // ---- 5) output projection ----
    gemm_bf16_wmma_kernel<8, 2><<<((NQ / 16) * 2 + 7) / 8, 256, 0, stream>>>(
        feats, pk_wout, b_out, out, NQ, 0);

    (void)in_sizes; (void)n_in; (void)out_size; (void)ws_size;
}
